// RowLSTM_44135083934045
// MI455X (gfx1250) — matrix operand order repaired and checked by reading, not compiled, not run on hardware
//
#include <hip/hip_runtime.h>
#include <hip/hip_bf16.h>

typedef __attribute__((ext_vector_type(16))) __bf16 v16bf;
typedef __attribute__((ext_vector_type(8)))  __bf16 v8bf;
typedef __attribute__((ext_vector_type(8)))  float  v8f;

#define N_LAYERS 2
#define HID 64
#define W32 32
#define H32 32
#define BATCH 64
#define NOUT 256      // 4*HID gates
#define LSTRIDE 33    // padded f32 LDS row stride (co-prime with 64 banks)
#define ASTR 264      // acts row stride in bf16 (multiple of 8 -> 16B aligned rows)

// ---------------- workspace layout (bytes) ----------------
// Wfrag  bf16 [2][16][8][32][16]           : 262144  @ 0
// biasC  f32  [2][256]                     : 2048    @ 262144
// WoutF  bf16 [48][2][32][16]              : 98304   @ 264192
// outh   bf16 [B*H*W][64]                  : 8388608 @ 362496
#define WS_WFRAG 0
#define WS_BIAS  262144
#define WS_WOUT  264192
#define WS_OUTH  362496

// Build a 16x32 bf16 A-fragment for one lane from a K-contiguous row:
// lane's elements are K = kc..kc+7 and K = kc+16..kc+23  (two b128 loads).
__device__ __forceinline__ v16bf ldA(const __bf16* __restrict__ row, int kc) {
    v8bf lo = *(const v8bf*)(row + kc);
    v8bf hi = *(const v8bf*)(row + kc + 16);
    v16bf r;
    #pragma unroll
    for (int j = 0; j < 8; ++j) { r[j] = lo[j]; r[j + 8] = hi[j]; }
    return r;
}

__device__ __forceinline__ float sigf(float x) { return 1.0f / (1.0f + __expf(-x)); }

// =====================================================================
// Kernel 1: repack weights into WMMA-B-fragment-linear bf16
// =====================================================================
__global__ void prep_weights(const float* __restrict__ wx,   // [2][256][64]
                             const float* __restrict__ bx,   // [2][256]
                             const float* __restrict__ wh,   // [2][256][64][3]
                             const float* __restrict__ bh,   // [2][256]
                             const float* __restrict__ out_w,// [768][64]
                             __bf16* __restrict__ Wfrag,
                             float*  __restrict__ biasC,
                             __bf16* __restrict__ WoutF) {
    int gid = blockIdx.x * blockDim.x + threadIdx.x;   // 8192 threads
    // ---- LSTM combined weight fragments: gid = ((l*16+nb)*8+kk)*32+lane
    {
        int lane = gid & 31;
        int kk   = (gid >> 5) & 7;
        int nb   = (gid >> 8) & 15;
        int l    = gid >> 12;                           // 0..1
        int N    = nb * 16 + (lane & 15);
        int kb   = kk * 32 + ((lane < 16) ? 0 : 8);
        #pragma unroll
        for (int j = 0; j < 16; ++j) {
            int K = kb + (j < 8 ? j : j + 8);
            float v;
            if (K < 64) {
                v = wx[(l * 256 + N) * 64 + K];
            } else {
                int c  = (K - 64) & 63;
                int dw = (K - 64) >> 6;                 // 0:w-1 1:w 2:w+1
                v = wh[((l * 256 + N) * 64 + c) * 3 + dw];
            }
            Wfrag[gid * 16 + j] = (__bf16)v;
        }
    }
    if (gid < 512) biasC[gid] = bx[gid] + bh[gid];
    // ---- output-projection fragments: gid = (nb*2+kk)*32+lane, 48*2*32=3072
    if (gid < 3072) {
        int lane = gid & 31;
        int kk   = (gid >> 5) & 1;
        int nb   = gid >> 6;
        int N    = nb * 16 + (lane & 15);               // o in [0,768)
        int kb   = kk * 32 + ((lane < 16) ? 0 : 8);
        #pragma unroll
        for (int j = 0; j < 16; ++j)
            WoutF[gid * 16 + j] = (__bf16)out_w[N * 64 + kb + (j < 8 ? j : j + 8)];
    }
}

// =====================================================================
// Kernel 2: RowLSTM scan — one workgroup per batch element.
// 512 threads = 16 waves; wave `wid` owns gate columns [wid*16, wid*16+16).
// Activations live in LDS as bf16 [row=w+1][K=256], K-contiguous, with halo
// rows 0 and 33 absorbing the w-1/w+1 conv-tap shifts (no boundary branches).
// =====================================================================
__global__ __launch_bounds__(512)
void rowlstm(const int*   __restrict__ x,      // [B][3][32][32]
             const float* __restrict__ pre_w,  // [64][3]
             const float* __restrict__ pre_b,  // [64]
             const __bf16* __restrict__ Wfrag,
             const float*  __restrict__ biasC,
             __bf16* __restrict__ outh) {      // [B*H*W][64]
    __shared__ __align__(32) __bf16 sW[131072];       // 256 KB weight fragments
    __shared__ float sBias[512];
    __shared__ __align__(16) __bf16 acts[34 * ASTR];  // 17952 B
    __shared__ float sC[HID * LSTRIDE];               // 8448 B (f32 cell state)
    __shared__ float sA[NOUT * LSTRIDE];              // 33792 B gate pre-acts

    const int b    = blockIdx.x;
    const int tid  = threadIdx.x;
    const int lane = tid & 31;
    const int wid  = tid >> 5;                        // 0..15 -> N tile
    const int mlo  = lane & 15;
    const int kbA  = (lane < 16) ? 0 : 8;             // A-frag K base
    const int wbase_lo = (lane >> 4) << 3;            // D rows: +0 / +8

    // ---- one-time staging: weights + bias to LDS; zero state & acts
    {
        const uint4* src = (const uint4*)Wfrag;
        uint4*       dst = (uint4*)sW;
        for (int i = tid; i < 16384; i += 512) dst[i] = src[i];
        if (tid < 512) sBias[tid] = biasC[tid];
        for (int i = tid; i < HID * LSTRIDE; i += 512) sC[i] = 0.0f;
        for (int i = tid; i < 34 * ASTR; i += 512) acts[i] = (__bf16)0.0f;
    }
    __syncthreads();

    const __bf16* rowA0 = &acts[(1 + mlo) * ASTR];        // M-tile 0: w = mlo
    const __bf16* rowA1 = &acts[(1 + 16 + mlo) * ASTR];   // M-tile 1: w = 16+mlo

    for (int row = 0; row < H32; ++row) {
        // ---- pre: 1x1 conv for this row, written straight into acts[K<64]
        for (int idx = tid; idx < HID * W32; idx += 512) {
            int cc = idx >> 5, w = idx & 31;
            float s = pre_b[cc];
            #pragma unroll
            for (int ci = 0; ci < 3; ++ci) {
                float xn = (float)x[((b * 3 + ci) * H32 + row) * W32 + w] * (1.0f / 255.0f) - 0.5f;
                s += pre_w[cc * 3 + ci] * xn;
            }
            acts[(1 + w) * ASTR + cc] = (__bf16)s;
        }
        __syncthreads();

        for (int l = 0; l < N_LAYERS; ++l) {
            // ---- GEMM: [32 x 256] = acts[32 x 256K] * W[256K x 256]
            v8f acc0 = {}; v8f acc1 = {};
            const int wslab = (l * 16 + wid) * 8;
            #pragma unroll
            for (int kk = 0; kk < 8; ++kk) {
                int kc = kk * 32 + kbA;
                v16bf bfrag = *(const v16bf*)&sW[(wslab + kk) * 512 + lane * 16];
                acc0 = __builtin_amdgcn_wmma_f32_16x16x32_bf16(
                           false, ldA(rowA0, kc), false, bfrag, (short)0, acc0, false, false);
                acc1 = __builtin_amdgcn_wmma_f32_16x16x32_bf16(
                           false, ldA(rowA1, kc), false, bfrag, (short)0, acc1, false, false);
            }
            // ---- spill D tiles (+bias) for cross-wave gate combine
            {
                int o = wid * 16 + mlo;
                float bias = sBias[l * 256 + o];
                #pragma unroll
                for (int r = 0; r < 8; ++r) {
                    sA[o * LSTRIDE + wbase_lo + r]      = acc0[r] + bias;
                    sA[o * LSTRIDE + 16 + wbase_lo + r] = acc1[r] + bias;
                }
            }
            __syncthreads();
            // ---- elementwise LSTM cell update; h fan-out to all K regions
            for (int idx = tid; idx < HID * W32; idx += 512) {
                int cc = idx >> 5, w = idx & 31;
                float gi = sigf(sA[(cc)        * LSTRIDE + w]);
                float gf = sigf(sA[(64  + cc)  * LSTRIDE + w]);
                float go = sigf(sA[(128 + cc)  * LSTRIDE + w]);
                float gg = tanhf(sA[(192 + cc) * LSTRIDE + w]);
                float cs = gf * sC[cc * LSTRIDE + w] + gi * gg;
                sC[cc * LSTRIDE + w] = cs;
                __bf16 hb = (__bf16)(go * tanhf(cs));
                acts[(1 + w) * ASTR + cc]        = hb;   // next layer's x input
                acts[(2 + w) * ASTR + 64  + cc]  = hb;   // conv tap w-1 (halo @33)
                acts[(1 + w) * ASTR + 128 + cc]  = hb;   // conv tap w
                acts[(0 + w) * ASTR + 192 + cc]  = hb;   // conv tap w+1 (halo @0)
                if (l == 1)                               // row output, [m][c] order
                    outh[(((b * H32 + row) * W32) + w) * 64 + cc] = hb;
            }
            __syncthreads();
        }
    }
}

// =====================================================================
// Kernel 3: output projection GEMM [65536 x 768] = [65536 x 64]*[64 x 768]
// + bias + scatter into [B, 3, H, W, 256] layout.
// =====================================================================
__global__ __launch_bounds__(256)
void outgemm(const __bf16* __restrict__ outh,
             const __bf16* __restrict__ WoutF,
             const float*  __restrict__ out_b,
             float* __restrict__ out) {
    const int m0   = blockIdx.x * 16;
    const int tid  = threadIdx.x;
    const int lane = tid & 31;
    const int wid  = tid >> 5;                 // 0..7
    const int mlo  = lane & 15;
    const int kbA  = (lane < 16) ? 0 : 8;
    const int m    = m0 + mlo;

    // A fragments for K = 0..31 and 32..63 (shared across all N tiles)
    const __bf16* arow = outh + m * 64;
    v16bf a0 = ldA(arow, kbA);
    v16bf a1 = ldA(arow, 32 + kbA);

    const int mb = m0 + ((lane >> 4) << 3);
    for (int nb = wid; nb < 48; nb += 8) {
        v16bf b0 = *(const v16bf*)&WoutF[(nb * 2 + 0) * 512 + lane * 16];
        v16bf b1 = *(const v16bf*)&WoutF[(nb * 2 + 1) * 512 + lane * 16];
        v8f acc = {};
        acc = __builtin_amdgcn_wmma_f32_16x16x32_bf16(false, a0, false, b0, (short)0, acc, false, false);
        acc = __builtin_amdgcn_wmma_f32_16x16x32_bf16(false, a1, false, b1, (short)0, acc, false, false);

        int o    = nb * 16 + mlo;              // 0..767
        float bs = out_b[o];
        int ci = o >> 8, p = o & 255;
        #pragma unroll
        for (int r = 0; r < 8; ++r) {
            int mm = mb + r;
            int bb = mm >> 10, hh = (mm >> 5) & 31, ww = mm & 31;
            out[((((bb * 3 + ci) * H32) + hh) * W32 + ww) * 256 + p] = acc[r] + bs;
        }
    }
}

// =====================================================================
extern "C" void kernel_launch(void* const* d_in, const int* in_sizes, int n_in,
                              void* d_out, int out_size, void* d_ws, size_t ws_size,
                              hipStream_t stream) {
    const int*   x     = (const int*)  d_in[0];
    const float* pre_w = (const float*)d_in[1];
    const float* pre_b = (const float*)d_in[2];
    const float* wx    = (const float*)d_in[3];
    const float* bx    = (const float*)d_in[4];
    const float* wh    = (const float*)d_in[5];
    const float* bh    = (const float*)d_in[6];
    const float* out_w = (const float*)d_in[7];
    const float* out_b = (const float*)d_in[8];
    float* out = (float*)d_out;

    char* ws = (char*)d_ws;
    __bf16* Wfrag = (__bf16*)(ws + WS_WFRAG);
    float*  biasC = (float*) (ws + WS_BIAS);
    __bf16* WoutF = (__bf16*)(ws + WS_WOUT);
    __bf16* outh  = (__bf16*)(ws + WS_OUTH);

    prep_weights<<<32, 256, 0, stream>>>(wx, bx, wh, bh, out_w, Wfrag, biasC, WoutF);
    rowlstm<<<BATCH, 512, 0, stream>>>(x, pre_w, pre_b, Wfrag, biasC, outh);
    outgemm<<<(BATCH * H32 * W32) / 16, 256, 0, stream>>>(outh, WoutF, out_b, out);
}
